// LSTMNet_65798898975059
// MI455X (gfx1250) — matrix operand order repaired and checked by reading, not compiled, not run on hardware
//
#include <hip/hip_runtime.h>
#include <hip/hip_bf16.h>
#include <math.h>

typedef __attribute__((ext_vector_type(16))) __bf16 v16bf;
typedef __attribute__((ext_vector_type(8)))  float  v8f;

#define HID 10
#define DIN 512
#define EPS 1e-5f
#define STATS_N 64
#define NPART_BLK 128

// ---------------------------------------------------------------------------
// Phase 1: y1 = x @ W1^T + b1   via v_wmma_f32_16x16x32_bf16
// One wave handles a 16-row tile of x; K loop 512/32 = 16 WMMAs.
// A (16x32 bf16): lane L<16 holds row M=L; element j -> K = half*8+j (j<8),
// 16+half*8+(j-8) (j>=8), half = lane>>4.
// B (32x16 bf16): lane holds column N = lane&15; element j -> same K as A.
// C/D (16x16 f32): lane = col N, vgpr v = row M = half*8+v.
// Out-of-range rows/cols are handled by ADDRESS CLAMPING, not predication:
// D[row][col] depends only on A[row] and B[col], and invalid rows/cols are
// never stored, so garbage there is harmless. This keeps EXEC all-ones and
// the inner loop branch-free (no saveexec / v_nop in the WMMA loop).
// ---------------------------------------------------------------------------
__global__ void gemm1_wmma(const float* __restrict__ x,
                           const float* __restrict__ W1,
                           const float* __restrict__ b1,
                           float* __restrict__ y1, int N) {
  const int lane = threadIdx.x & 31;
  const int wave = threadIdx.x >> 5;
  const int tileRow = (blockIdx.x * 8 + wave) * 16;
  const int half = lane >> 4;
  const int r    = lane & 15;

  int arow = tileRow + r;              // A row this lane feeds
  if (arow > N - 1) arow = N - 1;      // clamp: result rows >= N never stored
  const int bcol = (r < HID) ? r : (HID - 1);  // clamp: cols >= 10 never stored
  const float* xrow  = x  + (size_t)arow * DIN;
  const float* w1row = W1 + (size_t)bcol * DIN;

  v8f acc = {};
#pragma unroll 2
  for (int kt = 0; kt < DIN; kt += 32) {
    v16bf a, b;
    const float* pa0 = xrow + kt + half * 8;
    const float* pa1 = xrow + kt + 16 + half * 8;
#pragma unroll
    for (int j = 0; j < 8; ++j) { a[j] = (__bf16)pa0[j]; a[8 + j] = (__bf16)pa1[j]; }
    const float* pb0 = w1row + kt + half * 8;
    const float* pb1 = w1row + kt + 16 + half * 8;
#pragma unroll
    for (int j = 0; j < 8; ++j) { b[j] = (__bf16)pb0[j]; b[8 + j] = (__bf16)pb1[j]; }
    acc = __builtin_amdgcn_wmma_f32_16x16x32_bf16(false, a, false, b,
                                                  (short)0, acc, false, false);
  }
  if (r < HID) {
    const float bias = b1[r];
#pragma unroll
    for (int v = 0; v < 8; ++v) {
      const int t = tileRow + half * 8 + v;
      if (t < N) y1[(size_t)t * HID + r] = acc[v] + bias;
    }
  }
}

// ---------------------------------------------------------------------------
// Deterministic per-feature sum/sumsq: stage 1 (block tree-reduce -> partials)
// partials layout: block b -> part[b*32 + k] = sum_k, part[b*32+16+k] = sq_k
// ---------------------------------------------------------------------------
__global__ void bn_stats_part(const float* __restrict__ y, float* __restrict__ part, int N) {
  __shared__ float s[20][256];
  const int tid = threadIdx.x;
  float ls[HID], lq[HID];
#pragma unroll
  for (int k = 0; k < HID; ++k) { ls[k] = 0.f; lq[k] = 0.f; }
  for (int t = blockIdx.x * blockDim.x + tid; t < N; t += gridDim.x * blockDim.x) {
#pragma unroll
    for (int k = 0; k < HID; ++k) {
      const float v = y[(size_t)t * HID + k];
      ls[k] += v; lq[k] += v * v;
    }
  }
#pragma unroll
  for (int k = 0; k < HID; ++k) { s[k][tid] = ls[k]; s[10 + k][tid] = lq[k]; }
  __syncthreads();
  for (int off = 128; off > 0; off >>= 1) {
    if (tid < off)
#pragma unroll
      for (int k = 0; k < 20; ++k) s[k][tid] += s[k][tid + off];
    __syncthreads();
  }
  if (tid < HID) {
    part[blockIdx.x * 32 + tid]      = s[tid][0];
    part[blockIdx.x * 32 + 16 + tid] = s[10 + tid][0];
  }
}

// Stage 2: fixed-order final reduction (deterministic, no atomics)
__global__ void bn_stats_reduce(const float* __restrict__ part,
                                float* __restrict__ dsum, float* __restrict__ dsq,
                                int nblocks) {
  const int k = threadIdx.x;
  if (k < HID) {
    float a = 0.f, b = 0.f;
    for (int i = 0; i < nblocks; ++i) {
      a += part[i * 32 + k];
      b += part[i * 32 + 16 + k];
    }
    dsum[k] = a; dsq[k] = b;
  }
}

// ---------------------------------------------------------------------------
// Phase 2.5: pre[t][row] = sum_k W_ih[row][k]*BN1(y1[t][k]) + b_ih[row] + b_hh[row]
// plus per-step control word ctrl[t] = zp | (bnd<<2)
// ---------------------------------------------------------------------------
__global__ void pre_kernel(const float* __restrict__ y1,
                           const float* __restrict__ sum1, const float* __restrict__ sq1,
                           const float* __restrict__ g1, const float* __restrict__ be1,
                           const float* __restrict__ Wih,
                           const float* __restrict__ bih, const float* __restrict__ bhh,
                           const int* __restrict__ zero_pad, const int* __restrict__ broad_id,
                           float* __restrict__ pre, int* __restrict__ ctrl, int N) {
  const int idx = blockIdx.x * blockDim.x + threadIdx.x;
  if (idx >= N * 40) return;
  const int t = idx / 40;
  const int row = idx - t * 40;
  const float invN = 1.0f / (float)N;
  float acc = bih[row] + bhh[row];
#pragma unroll
  for (int k = 0; k < HID; ++k) {
    const float m  = sum1[k] * invN;
    const float v  = sq1[k] * invN - m * m;
    const float rs = rsqrtf(v + EPS);
    const float xn = (y1[(size_t)t * HID + k] - m) * rs * g1[k] + be1[k];
    acc += Wih[row * HID + k] * xn;
  }
  pre[idx] = acc;
  if (row == 0) {
    const int zp  = zero_pad[t];
    const int bnd = (t == N - 1) || (broad_id[t + 1] != broad_id[t]);
    ctrl[t] = (zp & 3) | (bnd << 2);
  }
}

// ---------------------------------------------------------------------------
// Phase 3: serial LSTM scan. One wave; lane u (<10) owns unit u:
// 4 W_hh gate rows + fused biases in VGPRs, c/sh/sc lane-local, h broadcast
// via 10 shfl (ds_bpermute). Double-buffered pre/ctrl loads + prefetch.
// ---------------------------------------------------------------------------
__device__ __forceinline__ float sigf(float x) { return 1.0f / (1.0f + __expf(-x)); }

__global__ void lstm_scan(const float* __restrict__ pre, const int* __restrict__ ctrl,
                          const float* __restrict__ Whh,
                          const float* __restrict__ bih, const float* __restrict__ bhh,
                          float* __restrict__ outs, int N) {
  const int lane = threadIdx.x;
  float w0[HID], w1[HID], w2[HID], w3[HID];
  float bs0 = 0.f, bs1 = 0.f, bs2 = 0.f, bs3 = 0.f;
  if (lane < HID) {
#pragma unroll
    for (int j = 0; j < HID; ++j) {
      w0[j] = Whh[(0 * HID + lane) * HID + j];
      w1[j] = Whh[(1 * HID + lane) * HID + j];
      w2[j] = Whh[(2 * HID + lane) * HID + j];
      w3[j] = Whh[(3 * HID + lane) * HID + j];
    }
    bs0 = bih[lane] + bhh[lane];
    bs1 = bih[HID + lane] + bhh[HID + lane];
    bs2 = bih[2 * HID + lane] + bhh[2 * HID + lane];
    bs3 = bih[3 * HID + lane] + bhh[3 * HID + lane];
  } else {
#pragma unroll
    for (int j = 0; j < HID; ++j) { w0[j] = w1[j] = w2[j] = w3[j] = 0.f; }
  }

  float hv[HID];
#pragma unroll
  for (int j = 0; j < HID; ++j) hv[j] = 0.f;
  float cst = 0.f, sh = 0.f, sc = 0.f, cnt = 0.f;

  // prime double buffer for t = 0
  float p0 = 0.f, p1 = 0.f, p2 = 0.f, p3 = 0.f;
  if (lane < HID) {
    p0 = pre[lane]; p1 = pre[HID + lane]; p2 = pre[2 * HID + lane]; p3 = pre[3 * HID + lane];
  }
  int cw = ctrl[0];

  for (int t = 0; t < N; ++t) {
    // issue next-step loads early (latency hidden behind cell compute)
    float n0 = 0.f, n1 = 0.f, n2 = 0.f, n3 = 0.f;
    int ncw = 0;
    if (t + 1 < N) {
      const float* pb = pre + (size_t)(t + 1) * 40;
      if (lane < HID) { n0 = pb[lane]; n1 = pb[HID + lane]; n2 = pb[2 * HID + lane]; n3 = pb[3 * HID + lane]; }
      ncw = ctrl[t + 1];
    }
    if (t + 8 < N) __builtin_prefetch(pre + (size_t)(t + 8) * 40, 0, 0);

    // zero-input pad cells: advance (h, c) exactly zp times
    const int zp = cw & 3;
    for (int k = 0; k < zp; ++k) {
      float m0 = 0.f, m1 = 0.f, m2 = 0.f, m3 = 0.f;
#pragma unroll
      for (int j = 0; j < HID; ++j) {
        m0 += w0[j] * hv[j]; m1 += w1[j] * hv[j];
        m2 += w2[j] * hv[j]; m3 += w3[j] * hv[j];
      }
      const float gi = sigf(bs0 + m0), gf = sigf(bs1 + m1);
      const float gg = tanhf(bs2 + m2), go = sigf(bs3 + m3);
      cst = gf * cst + gi * gg;
      const float h2 = go * tanhf(cst);
#pragma unroll
      for (int j = 0; j < HID; ++j) hv[j] = __shfl(h2, j);
    }

    // main cell (does NOT advance h/c; accumulates group sums)
    {
      float m0 = 0.f, m1 = 0.f, m2 = 0.f, m3 = 0.f;
#pragma unroll
      for (int j = 0; j < HID; ++j) {
        m0 += w0[j] * hv[j]; m1 += w1[j] * hv[j];
        m2 += w2[j] * hv[j]; m3 += w3[j] * hv[j];
      }
      const float gi = sigf(p0 + m0), gf = sigf(p1 + m1);
      const float gg = tanhf(p2 + m2), go = sigf(p3 + m3);
      const float ct2 = gf * cst + gi * gg;
      const float ht  = go * tanhf(ct2);
      sh += ht; sc += ct2; cnt += 1.0f;
      if (lane < HID) outs[(size_t)t * HID + lane] = ht;
      if (cw & 4) {  // group boundary: h,c <- group means; reset sums
        const float hn = sh / cnt;
        cst = sc / cnt;
#pragma unroll
        for (int j = 0; j < HID; ++j) hv[j] = __shfl(hn, j);
        sh = 0.f; sc = 0.f; cnt = 0.f;
      }
    }

    p0 = n0; p1 = n1; p2 = n2; p3 = n3; cw = ncw;
  }
}

// ---------------------------------------------------------------------------
// Phase 4: out[t] = tanh( BN2(outs[t]) . W2 + b2 )
// ---------------------------------------------------------------------------
__global__ void final_kernel(const float* __restrict__ outs,
                             const float* __restrict__ sum2, const float* __restrict__ sq2,
                             const float* __restrict__ g2, const float* __restrict__ be2,
                             const float* __restrict__ W2, const float* __restrict__ b2,
                             float* __restrict__ out, int N) {
  const int t = blockIdx.x * blockDim.x + threadIdx.x;
  if (t >= N) return;
  const float invN = 1.0f / (float)N;
  float acc = b2[0];
#pragma unroll
  for (int k = 0; k < HID; ++k) {
    const float m  = sum2[k] * invN;
    const float v  = sq2[k] * invN - m * m;
    const float rs = rsqrtf(v + EPS);
    const float xn = (outs[(size_t)t * HID + k] - m) * rs * g2[k] + be2[k];
    acc += W2[k] * xn;
  }
  out[t] = tanhf(acc);
}

// ---------------------------------------------------------------------------
extern "C" void kernel_launch(void* const* d_in, const int* in_sizes, int n_in,
                              void* d_out, int out_size, void* d_ws, size_t ws_size,
                              hipStream_t stream) {
  const float* x        = (const float*)d_in[0];
  const int*   zero_pad = (const int*)  d_in[1];
  const int*   broad_id = (const int*)  d_in[2];
  const float* W1       = (const float*)d_in[3];
  const float* b1       = (const float*)d_in[4];
  const float* g1       = (const float*)d_in[5];
  const float* be1      = (const float*)d_in[6];
  const float* W_ih     = (const float*)d_in[7];
  const float* W_hh     = (const float*)d_in[8];
  const float* b_ih     = (const float*)d_in[9];
  const float* b_hh     = (const float*)d_in[10];
  const float* g2       = (const float*)d_in[11];
  const float* be2      = (const float*)d_in[12];
  const float* W2       = (const float*)d_in[13];
  const float* b2       = (const float*)d_in[14];
  float* out = (float*)d_out;

  const int N = in_sizes[0] / DIN;  // 65536

  // workspace layout (floats)
  float* ws    = (float*)d_ws;
  float* sum1  = ws + 0;
  float* sq1   = ws + 16;
  float* sum2  = ws + 32;
  float* sq2   = ws + 48;
  float* part  = ws + STATS_N;                   // NPART_BLK * 32
  float* y1    = part + NPART_BLK * 32;          // N*10
  float* pre   = y1 + (size_t)N * HID;           // N*40
  float* outs  = pre + (size_t)N * 40;           // N*10
  int*   ctrl  = (int*)(outs + (size_t)N * HID); // N

  // Phase 1: GEMM via WMMA (8 waves/block, 16 rows/wave)
  const int tiles  = (N + 15) / 16;
  const int blocks = (tiles + 7) / 8;
  gemm1_wmma<<<blocks, 256, 0, stream>>>(x, W1, b1, y1, N);

  // Phase 2: BN1 stats (deterministic two-stage)
  bn_stats_part<<<NPART_BLK, 256, 0, stream>>>(y1, part, N);
  bn_stats_reduce<<<1, 32, 0, stream>>>(part, sum1, sq1, NPART_BLK);

  // Phase 2.5: input-gate precompute + control words
  pre_kernel<<<(N * 40 + 255) / 256, 256, 0, stream>>>(
      y1, sum1, sq1, g1, be1, W_ih, b_ih, b_hh, zero_pad, broad_id, pre, ctrl, N);

  // Phase 3: serial scan, one wave
  lstm_scan<<<1, 32, 0, stream>>>(pre, ctrl, W_hh, b_ih, b_hh, outs, N);

  // Phase 4: BN2 stats + final projection
  bn_stats_part<<<NPART_BLK, 256, 0, stream>>>(outs, part, N);
  bn_stats_reduce<<<1, 32, 0, stream>>>(part, sum2, sq2, NPART_BLK);
  final_kernel<<<(N + 255) / 256, 256, 0, stream>>>(outs, sum2, sq2, g2, be2, W2, b2, out, N);
}
